// UnidirectionalMultiheadAttention_46231027974720
// MI455X (gfx1250) — matrix operand order repaired and checked by reading, not compiled, not run on hardware
//
#include <hip/hip_runtime.h>
#include <hip/hip_bf16.h>

typedef __bf16 bf16_t;
typedef bf16_t v16bf __attribute__((ext_vector_type(16)));
typedef bf16_t v8bf  __attribute__((ext_vector_type(8)));
typedef bf16_t v4bf  __attribute__((ext_vector_type(4)));
typedef float  v8f   __attribute__((ext_vector_type(8)));
typedef float  v4f   __attribute__((ext_vector_type(4)));

typedef unsigned int uint32x4 __attribute__((ext_vector_type(4)));
typedef int          int32x4  __attribute__((ext_vector_type(4)));
typedef int          int32x8  __attribute__((ext_vector_type(8)));

#define HIDDEN 2048
#define NHEADS 16
#define HD     128
#define BATCH  2
#define SEQ    2048
#define MROWS  (BATCH * SEQ)

#if defined(__has_builtin)
#  if __has_builtin(__builtin_amdgcn_tensor_load_to_lds)
#    define HAVE_TDM 1
#  endif
#endif
#ifndef HAVE_TDM
#  define HAVE_TDM 0
#endif

// ---------------- Tensor Data Mover: 2-D bf16 tile -> LDS (ISA ch.8 D#) ----
// Builds a D# for a 2-D tile of 2-byte elements inside a row-major tensor and
// issues TENSOR_LOAD_TO_LDS. LDS destination rows can be padded via the TDM
// pad engine (pad_interval/pad_amount codes per D# group1). Descriptors are
// uniform; TDM ignores EXEC, so we issue from a single wave and track with
// TENSORcnt.
#if HAVE_TDM
__device__ __forceinline__ void tdm_load_2d_bf16(
    unsigned lds_addr, const void* gptr,
    unsigned tile_x, unsigned tile_y,        // tile dims (elements)
    unsigned tensor_x, unsigned tensor_y,    // tensor dims (elements)
    unsigned stride_x,                       // tensor dim0 stride (elements)
    unsigned pad_interval_code, unsigned pad_amount_code)
{
    unsigned long long ga = (unsigned long long)gptr;
    uint32x4 g0;
    g0[0] = 1u;                                    // count=1 (valid), user mode
    g0[1] = lds_addr;                              // [63:32] lds_addr
    g0[2] = (unsigned)ga;                          // [95:64] global_addr lo
    g0[3] = (unsigned)((ga >> 32) & 0x01FFFFFFu)   // [120:96] global_addr hi
          | (2u << 30);                            // [127:126] type=2 (image)
    int32x8 g1;
    g1[0] = (int)((1u << 16)                       // data_size = 2 bytes
          | (1u << 20)                             // pad_enable
          | (pad_interval_code << 22)
          | (pad_amount_code << 25));
    g1[1] = (int)(tensor_x << 16);                 // [79:48] tensor_dim0 (lo16)
    g1[2] = (int)((tensor_x >> 16) | (tensor_y << 16)); // dim0 hi / dim1 lo
    g1[3] = (int)((tensor_y >> 16) | (tile_x << 16));   // dim1 hi / tile_dim0
    g1[4] = (int)tile_y;                           // tile_dim1 (tile_dim2 = 0)
    g1[5] = (int)stride_x;                         // tensor_dim0_stride lo32
    g1[6] = 0;                                     // stride hi, dim1_stride lo
    g1[7] = 0;
    int32x4 gz = {};                               // groups 2/3: unused (2-D)
#if __clang_major__ >= 23
    int32x8 gz8 = {};
    __builtin_amdgcn_tensor_load_to_lds(g0, g1, gz, gz, gz8, 0);
#else
    __builtin_amdgcn_tensor_load_to_lds(g0, g1, gz, gz, 0);
#endif
}
#endif

// ---------------- WMMA fragment loaders (wave32, ISA 7.12.2 layouts) -------

// A matrix 16x32 bf16, source is row-major M x K with leading dim `ld`.
__device__ __forceinline__ v16bf load_a16x32(const bf16_t* base, int ld, int lane) {
    int m = lane & 15, h = (lane >> 4) & 1;
    const bf16_t* p = base + m * ld + 8 * h;
    union { v16bf v; v8bf h8[2]; } u;
    u.h8[0] = *(const v8bf*)(p);        // K = 8h .. 8h+7
    u.h8[1] = *(const v8bf*)(p + 16);   // K = 8h+16 .. 8h+23
    return u.v;
}

// B matrix 32x16 bf16, source stored TRANSPOSED as N x K row-major.
__device__ __forceinline__ v16bf load_b32x16(const bf16_t* base, int ld, int lane) {
    int n = lane & 15, h = (lane >> 4) & 1;
    const bf16_t* p = base + n * ld + 8 * h;
    union { v16bf v; v8bf h8[2]; } u;
    u.h8[0] = *(const v8bf*)(p);
    u.h8[1] = *(const v8bf*)(p + 16);
    return u.v;
}

#define WMMA_BF16(A, B, C) \
    __builtin_amdgcn_wmma_f32_16x16x32_bf16(false, (A), false, (B), (short)0, (C), false, false)

// ---------------- Kernel 1: QKV projection -------------------------------

#define BM   64
#define BN   64
#define BK   32
#define KPAD 40   // 80B LDS rows: 16B aligned, low bank conflict

__global__ __launch_bounds__(128) void qkv_gemm_kernel(
    const float* __restrict__ x, const float* __restrict__ Wqkv,
    const float* __restrict__ bqkv,
    bf16_t* __restrict__ Qb, bf16_t* __restrict__ Kb, bf16_t* __restrict__ Vb)
{
    __shared__ bf16_t As[BM * KPAD];   // M x K
    __shared__ bf16_t Bs[BN * KPAD];   // N x K (transposed while staging)

    const int tid  = threadIdx.x;
    const int lane = tid & 31;
    const int wave = tid >> 5;
    const int m0 = blockIdx.y * BM;
    const int n0 = blockIdx.x * BN;
    const int wm = (wave >> 1) * 32;
    const int wn = (wave & 1) * 32;

    v8f acc[2][2] = {};

    for (int k0 = 0; k0 < HIDDEN; k0 += BK) {
        // prefetch next k-tile of the streamed weight into L2
        if (k0 + BK < HIDDEN) {
            __builtin_prefetch(Wqkv + (size_t)(k0 + BK + (tid >> 2)) * (3 * HIDDEN)
                                    + n0 + (tid & 3) * 16, 0, 1);
        }
        __syncthreads();
        // stage A: 64x32 fp32 -> bf16
        #pragma unroll
        for (int i = 0; i < 4; ++i) {
            int idx = tid + i * 128;
            int row = idx >> 3;
            int c4  = idx & 7;
            v4f v = *(const v4f*)(x + (size_t)(m0 + row) * HIDDEN + k0 + c4 * 4);
            v4bf bv;
            bv[0] = (bf16_t)v[0]; bv[1] = (bf16_t)v[1];
            bv[2] = (bf16_t)v[2]; bv[3] = (bf16_t)v[3];
            *(v4bf*)(As + row * KPAD + c4 * 4) = bv;
        }
        // stage B transposed: Wqkv[k][n] -> Bs[n][k]
        #pragma unroll
        for (int i = 0; i < 4; ++i) {
            int idx = tid + i * 128;
            int kr = idx >> 4;
            int c4 = idx & 15;
            v4f v = *(const v4f*)(Wqkv + (size_t)(k0 + kr) * (3 * HIDDEN) + n0 + c4 * 4);
            #pragma unroll
            for (int j = 0; j < 4; ++j)
                Bs[(c4 * 4 + j) * KPAD + kr] = (bf16_t)v[j];
        }
        __syncthreads();

        v16bf a0 = load_a16x32(As + (wm +  0) * KPAD, KPAD, lane);
        v16bf a1 = load_a16x32(As + (wm + 16) * KPAD, KPAD, lane);
        v16bf b0 = load_b32x16(Bs + (wn +  0) * KPAD, KPAD, lane);
        v16bf b1 = load_b32x16(Bs + (wn + 16) * KPAD, KPAD, lane);
        acc[0][0] = WMMA_BF16(a0, b0, acc[0][0]);
        acc[0][1] = WMMA_BF16(a0, b1, acc[0][1]);
        acc[1][0] = WMMA_BF16(a1, b0, acc[1][0]);
        acc[1][1] = WMMA_BF16(a1, b1, acc[1][1]);
    }

    // epilogue: bias + scatter to Q/K/V bf16 [B][NH][S][HD]
    const int lm = lane & 15, lh = lane >> 4;
    #pragma unroll
    for (int i = 0; i < 2; ++i) {
        #pragma unroll
        for (int j = 0; j < 2; ++j) {
            int n = n0 + wn + j * 16 + lm;
            float bias = bqkv[n];
            int c      = n >> 11;        // 0 = Q, 1 = K, 2 = V
            int within = n & 2047;
            int head   = within >> 7;
            int hd     = within & 127;
            bf16_t* dstbuf = (c == 0) ? Qb : (c == 1) ? Kb : Vb;
            #pragma unroll
            for (int r = 0; r < 8; ++r) {
                int m = m0 + wm + i * 16 + r + 8 * lh;
                int b = m / SEQ;
                int s = m & (SEQ - 1);
                size_t dst = ((size_t)(b * NHEADS + head) * SEQ + s) * HD + hd;
                dstbuf[dst] = (bf16_t)(acc[i][j][r] + bias);
            }
        }
    }
}

// ---------------- Kernel 2: causal flash attention ------------------------

#define HDP 136   // 272B K rows in LDS (TDM pad: 64DW interval + 4DW pad)
#define KTP 40

__global__ __launch_bounds__(128) void flash_attn_kernel(
    const bf16_t* __restrict__ Qb, const bf16_t* __restrict__ Kb,
    const bf16_t* __restrict__ Vb, bf16_t* __restrict__ Ao)
{
    __shared__ bf16_t Ks [32 * HDP];      // [kk][hd]  (= N x K for Q*K^T)
    __shared__ bf16_t Vts[HD * KTP];      // [hd][kk]  (= N x K for P*V)
    __shared__ bf16_t Ps [4 * 16 * KTP];  // per-wave P tile, [q][kk]

    const int tid   = threadIdx.x;
    const int lane  = tid & 31;
    const int wave  = tid >> 5;
    const int lm    = lane & 15, lh = lane >> 4;
    const int bh    = blockIdx.y;            // b*NH + head
    const int qbase = blockIdx.x * 64;
    const int qw    = qbase + wave * 16;

    const size_t hoff = (size_t)bh * SEQ * HD;
    const bf16_t* Qh = Qb + hoff;
    const bf16_t* Kh = Kb + hoff;
    const bf16_t* Vh = Vb + hoff;

    v16bf qa[4];
    #pragma unroll
    for (int c = 0; c < 4; ++c)
        qa[c] = load_a16x32(Qh + (size_t)qw * HD + c * 32, HD, lane);

    v8f o[8] = {};
    float mrow[8], lrow[8];
    #pragma unroll
    for (int r = 0; r < 8; ++r) { mrow[r] = -3.0e38f; lrow[r] = 0.0f; }

    const float scale = 0.08838834764831845f;   // 1/sqrt(128)
    const int nkb = blockIdx.x * 2 + 2;          // causal bound

    for (int kb = 0; kb < nkb; ++kb) {
        const int g0 = kb * 32;
        __syncthreads();
#if HAVE_TDM
        // K tile (32 x 128 bf16) via the Tensor Data Mover, with the TDM pad
        // engine inserting 16B after every 256B row (128 -> 136 elem rows).
        if (wave == 0) {
            tdm_load_2d_bf16((unsigned)(size_t)&Ks[0], Kh + (size_t)g0 * HD,
                             /*tile*/ HD, 32, /*tensor*/ HD, SEQ,
                             /*stride*/ HD, /*pad_int 64DW*/ 5, /*pad 4DW*/ 3);
        }
#endif
        // stage V transposed (and K manually if no TDM)
        #pragma unroll
        for (int i = 0; i < 4; ++i) {
            int idx = tid + i * 128;             // 512 chunks of 8 bf16
            int kk  = idx >> 4;
            int c8  = (idx & 15) * 8;
#if !HAVE_TDM
            v8bf kv = *(const v8bf*)(Kh + (size_t)(g0 + kk) * HD + c8);
            *(v8bf*)(Ks + kk * HDP + c8) = kv;
#endif
            v8bf vv = *(const v8bf*)(Vh + (size_t)(g0 + kk) * HD + c8);
            #pragma unroll
            for (int j = 0; j < 8; ++j)
                Vts[(c8 + j) * KTP + kk] = vv[j];
        }
#if HAVE_TDM
        if (wave == 0) __builtin_amdgcn_s_wait_tensorcnt(0);
#endif
        __syncthreads();

        // scores: S = Q * K^T, 16 queries x 32 keys
        v8f s0 = {}, s1 = {};
        #pragma unroll
        for (int c = 0; c < 4; ++c) {
            v16bf b0 = load_b32x16(Ks +  0 * HDP + c * 32, HDP, lane);
            v16bf b1 = load_b32x16(Ks + 16 * HDP + c * 32, HDP, lane);
            s0 = WMMA_BF16(qa[c], b0, s0);
            s1 = WMMA_BF16(qa[c], b1, s1);
        }

        bf16_t* Pw = Ps + wave * 16 * KTP;
        #pragma unroll
        for (int r = 0; r < 8; ++r) {
            const int qrow = qw + r + 8 * lh;
            float v0 = s0[r] * scale, v1 = s1[r] * scale;
            if (g0 + lm      > qrow) v0 = -3.0e38f;   // causal mask
            if (g0 + 16 + lm > qrow) v1 = -3.0e38f;
            float mx = fmaxf(v0, v1);
            #pragma unroll
            for (int off = 1; off < 16; off <<= 1)
                mx = fmaxf(mx, __shfl_xor(mx, off, 32));
            float mnew  = fmaxf(mrow[r], mx);
            float alpha = __expf(mrow[r] - mnew);
            mrow[r] = mnew;
            float p0 = __expf(v0 - mnew);
            float p1 = __expf(v1 - mnew);
            float rs = p0 + p1;
            #pragma unroll
            for (int off = 1; off < 16; off <<= 1)
                rs += __shfl_xor(rs, off, 32);
            lrow[r] = lrow[r] * alpha + rs;
            #pragma unroll
            for (int t = 0; t < 8; ++t) o[t][r] *= alpha;
            Pw[(r + 8 * lh) * KTP + lm]      = (bf16_t)p0;   // D -> A layout
            Pw[(r + 8 * lh) * KTP + 16 + lm] = (bf16_t)p1;
        }

        // O += P * V
        v16bf pa = load_a16x32(Pw, KTP, lane);
        #pragma unroll
        for (int t = 0; t < 8; ++t) {
            v16bf bv = load_b32x16(Vts + (t * 16) * KTP, KTP, lane);
            o[t] = WMMA_BF16(pa, bv, o[t]);
        }
    }

    // epilogue: normalize, store bf16 to Ao[b][s][head*HD + hd]
    const int b = bh / NHEADS;
    const int head = bh % NHEADS;
    #pragma unroll
    for (int r = 0; r < 8; ++r) {
        float inv = 1.0f / lrow[r];
        int srow = qw + r + 8 * lh;
        size_t base = ((size_t)b * SEQ + srow) * HIDDEN + head * HD;
        #pragma unroll
        for (int t = 0; t < 8; ++t)
            Ao[base + t * 16 + lm] = (bf16_t)(o[t][r] * inv);
    }
}

// ---------------- Kernel 3: output projection -----------------------------

__global__ __launch_bounds__(128) void out_gemm_kernel(
    const bf16_t* __restrict__ Ao, const float* __restrict__ Wo,
    const float* __restrict__ bo, float* __restrict__ out)
{
    __shared__ bf16_t As[BM * KPAD];
    __shared__ bf16_t Bs[BN * KPAD];

    const int tid  = threadIdx.x;
    const int lane = tid & 31;
    const int wave = tid >> 5;
    const int m0 = blockIdx.y * BM;
    const int n0 = blockIdx.x * BN;
    const int wm = (wave >> 1) * 32;
    const int wn = (wave & 1) * 32;

    v8f acc[2][2] = {};

    for (int k0 = 0; k0 < HIDDEN; k0 += BK) {
        if (k0 + BK < HIDDEN) {
            __builtin_prefetch(Wo + (size_t)(k0 + BK + (tid >> 2)) * HIDDEN
                                  + n0 + (tid & 3) * 16, 0, 1);
        }
        __syncthreads();
#if HAVE_TDM
        // A tile (64 x 32 bf16) via TDM: 64B rows padded by 16B (32 -> 40 elems)
        if (wave == 0) {
            tdm_load_2d_bf16((unsigned)(size_t)&As[0],
                             Ao + (size_t)m0 * HIDDEN + k0,
                             /*tile*/ BK, BM, /*tensor*/ HIDDEN, MROWS,
                             /*stride*/ HIDDEN, /*pad_int 16DW*/ 3, /*pad 4DW*/ 3);
        }
#else
        #pragma unroll
        for (int i = 0; i < 2; ++i) {
            int idx = tid + i * 128;
            int row = idx >> 2;
            int c8  = (idx & 3) * 8;
            *(v8bf*)(As + row * KPAD + c8) =
                *(const v8bf*)(Ao + (size_t)(m0 + row) * HIDDEN + k0 + c8);
        }
#endif
        // stage B transposed: Wo[k][n] fp32 -> Bs[n][k] bf16
        #pragma unroll
        for (int i = 0; i < 4; ++i) {
            int idx = tid + i * 128;
            int kr = idx >> 4;
            int c4 = idx & 15;
            v4f v = *(const v4f*)(Wo + (size_t)(k0 + kr) * HIDDEN + n0 + c4 * 4);
            #pragma unroll
            for (int j = 0; j < 4; ++j)
                Bs[(c4 * 4 + j) * KPAD + kr] = (bf16_t)v[j];
        }
#if HAVE_TDM
        if (wave == 0) __builtin_amdgcn_s_wait_tensorcnt(0);
#endif
        __syncthreads();

        v16bf a0 = load_a16x32(As + (wm +  0) * KPAD, KPAD, lane);
        v16bf a1 = load_a16x32(As + (wm + 16) * KPAD, KPAD, lane);
        v16bf b0 = load_b32x16(Bs + (wn +  0) * KPAD, KPAD, lane);
        v16bf b1 = load_b32x16(Bs + (wn + 16) * KPAD, KPAD, lane);
        acc[0][0] = WMMA_BF16(a0, b0, acc[0][0]);
        acc[0][1] = WMMA_BF16(a0, b1, acc[0][1]);
        acc[1][0] = WMMA_BF16(a1, b0, acc[1][0]);
        acc[1][1] = WMMA_BF16(a1, b1, acc[1][1]);
    }

    const int lm = lane & 15, lh = lane >> 4;
    #pragma unroll
    for (int i = 0; i < 2; ++i) {
        #pragma unroll
        for (int j = 0; j < 2; ++j) {
            int n = n0 + wn + j * 16 + lm;
            float bias = bo[n];
            #pragma unroll
            for (int r = 0; r < 8; ++r) {
                int m = m0 + wm + i * 16 + r + 8 * lh;
                out[(size_t)m * HIDDEN + n] = acc[i][j][r] + bias;
            }
        }
    }
}

// ---------------- launch --------------------------------------------------

extern "C" void kernel_launch(void* const* d_in, const int* in_sizes, int n_in,
                              void* d_out, int out_size, void* d_ws, size_t ws_size,
                              hipStream_t stream) {
    const float* x    = (const float*)d_in[0];
    const float* Wqkv = (const float*)d_in[1];
    const float* bqkv = (const float*)d_in[2];
    const float* Wo   = (const float*)d_in[3];
    const float* bo   = (const float*)d_in[4];
    float* out = (float*)d_out;

    const size_t per = (size_t)BATCH * NHEADS * SEQ * HD;   // 8.39M elems
    bf16_t* Qb = (bf16_t*)d_ws;
    bf16_t* Kb = Qb + per;
    bf16_t* Vb = Kb + per;
    bf16_t* Ao = Vb + per;   // [B][S][HIDDEN] bf16

    qkv_gemm_kernel<<<dim3((3 * HIDDEN) / BN, MROWS / BM), 128, 0, stream>>>(
        x, Wqkv, bqkv, Qb, Kb, Vb);
    flash_attn_kernel<<<dim3(SEQ / 64, BATCH * NHEADS), 128, 0, stream>>>(
        Qb, Kb, Vb, Ao);
    out_gemm_kernel<<<dim3(HIDDEN / BN, MROWS / BM), 128, 0, stream>>>(
        Ao, Wo, bo, out);
}
